// GroupConv_67723044323985
// MI455X (gfx1250) — matrix operand order repaired and checked by reading, not compiled, not run on hardware
//
#include <hip/hip_runtime.h>
#include <hip/hip_bf16.h>

// ---------------------------------------------------------------------------
// Grouped 3x3 conv (NHWC, SAME) as implicit GEMM on CDNA5 WMMA bf16.
//   x      : [16,56,56,256] f32
//   kernels: [8,3,3,32,64]  f32   (flat per group/tap: [t][k=32][n=64])
//   bias   : [512]          f32
//   out    : [16,56,56,512] f32
//
// Block = 128 threads (4 waves), one group. Weights for the group are staged
// once into LDS in WMMA B-fragment order; each wave then sweeps 4 M-tiles of
// 16 spatial positions x 64 filters (9 taps x 4 N-tiles = 36 bf16 WMMAs per
// tile), amortizing the LDS fill 4x.
// ---------------------------------------------------------------------------

typedef __attribute__((ext_vector_type(16))) __bf16 v16bf;
typedef __attribute__((ext_vector_type(8)))  float  v8f;

#define NG     8
#define PCH    32
#define FCH    64
#define CINX   256
#define COUTX  512
#define HWDIM  56
#define NPOS   (16 * 56 * 56)          // 50176 spatial positions
#define MTILES (NPOS / 16)             // 3136
#define MT_PER_WAVE 4
#define WAVES_PER_BLOCK 4
#define MT_PER_BLOCK (MT_PER_WAVE * WAVES_PER_BLOCK)   // 16
#define WLDS_ELEMS (9 * 4 * 32 * 16)   // taps * ntiles * lanes * 16 bf16

__device__ __forceinline__ __bf16 f2bf(float f) {
    return static_cast<__bf16>(f);     // backend: native bf16 cvt (RNE)
}

__global__ __launch_bounds__(128)
void group_conv_wmma_bf16(const float* __restrict__ x,
                          const float* __restrict__ kern,
                          const float* __restrict__ bias,
                          float* __restrict__ out)
{
    // B-fragment layout: wlds[((t*4 + j)*32 + lane)*16 + e]
    //   element e of lane l  <->  k = e<8 ? (l>>4)*8+e : 16+(l>>4)*8+e-8 , n = j*16 + (l&15)
    __shared__ __align__(32) __bf16 wlds[WLDS_ELEMS];

    const int tid = threadIdx.x;
    const int g   = blockIdx.y;

    // ---- weight fill: quad-wise (4 consecutive filters n share one float4).
    // quad q: n4 = q[3:0] (n0 = n4*4), k = q[8:4], t = q[>=9].
    // kern flat offset needs no kh/kw split: (g*9 + t)*2048 + k*64 + n.
    for (int q = tid; q < 9 * 32 * 16; q += 128) {
        const int n4 = q & 15;
        const int k  = (q >> 4) & 31;
        const int t  = q >> 9;
        const int n0 = n4 << 2;
        const int j  = n4 >> 2;
        const int l0 = (n4 & 3) << 2;      // first of 4 consecutive l16 slots
        const int half = (k >> 3) & 1;
        const int e    = (k & 7) | ((k >> 4) << 3);

        const float4 f =
            *(const float4*)&kern[(g * 9 + t) * (PCH * FCH) + k * FCH + n0];

        const int base = (((t * 4 + j) * 32) + (half << 4) + l0) * 16 + e;
        wlds[base +  0] = f2bf(f.x);       // lane l0+0
        wlds[base + 16] = f2bf(f.y);       // lane l0+1
        wlds[base + 32] = f2bf(f.z);       // lane l0+2
        wlds[base + 48] = f2bf(f.w);       // lane l0+3
    }
    __syncthreads();

    const int wave = tid >> 5;
    const int lane = tid & 31;
    const int half = lane >> 4;     // K-half selector for A fragment
    const int l16  = lane & 15;     // A row (M) / B,C column (N)

    const int cbase = g * FCH;
    float bv[4];
    #pragma unroll
    for (int j = 0; j < 4; ++j) bv[j] = bias[cbase + j * 16 + l16];

    #pragma unroll 1
    for (int mt = 0; mt < MT_PER_WAVE; ++mt) {
        const int mtile = blockIdx.x * MT_PER_BLOCK + mt * WAVES_PER_BLOCK + wave;
        const int pos   = mtile * 16 + l16;      // this lane's A-row position
        const int hw    = pos % 3136;
        const int h     = hw / HWDIM;
        const int w     = hw - h * HWDIM;

        // Lane's A channels: two contiguous 8-float runs at (half*8) and
        // (16 + half*8) within the group's 32 channels.
        const float* xbase = x + (size_t)pos * CINX + g * PCH + half * 8;

        v8f acc[4] = {v8f{}, v8f{}, v8f{}, v8f{}};

        #pragma unroll
        for (int t = 0; t < 9; ++t) {
            const int kh = t / 3, kw = t - kh * 3;   // constants (unrolled)
            const int ih = h + kh - 1, iw = w + kw - 1;
            const bool valid = ((unsigned)ih < (unsigned)HWDIM) &&
                               ((unsigned)iw < (unsigned)HWDIM);
            const float* pa = xbase + ((kh - 1) * HWDIM + (kw - 1)) * CINX;

            float4 f0 = {0.f,0.f,0.f,0.f}, f1 = {0.f,0.f,0.f,0.f};
            float4 f2 = {0.f,0.f,0.f,0.f}, f3 = {0.f,0.f,0.f,0.f};
            if (valid) {
                f0 = *(const float4*)(pa);
                f1 = *(const float4*)(pa + 4);
                f2 = *(const float4*)(pa + 16);
                f3 = *(const float4*)(pa + 20);
            }

            // 16-bit A layout: e<8 -> k = half*8+e ; e>=8 -> k = 16+half*8+e-8
            v16bf a = {};
            a[0]  = f2bf(f0.x); a[1]  = f2bf(f0.y);
            a[2]  = f2bf(f0.z); a[3]  = f2bf(f0.w);
            a[4]  = f2bf(f1.x); a[5]  = f2bf(f1.y);
            a[6]  = f2bf(f1.z); a[7]  = f2bf(f1.w);
            a[8]  = f2bf(f2.x); a[9]  = f2bf(f2.y);
            a[10] = f2bf(f2.z); a[11] = f2bf(f2.w);
            a[12] = f2bf(f3.x); a[13] = f2bf(f3.y);
            a[14] = f2bf(f3.z); a[15] = f2bf(f3.w);

            #pragma unroll
            for (int j = 0; j < 4; ++j) {
                const v16bf b =
                    *(const v16bf*)&wlds[(((t * 4 + j) * 32) + lane) * 16];
                acc[j] = __builtin_amdgcn_wmma_f32_16x16x32_bf16(
                    false, a, false, b, (short)0, acc[j], false, false);
            }
        }

        // C/D f32 layout: lane -> N = l16, VGPR r -> M = r + half*8.
        #pragma unroll
        for (int j = 0; j < 4; ++j) {
            const int n = cbase + j * 16 + l16;
            #pragma unroll
            for (int r = 0; r < 8; ++r) {
                const int m = r + half * 8;
                out[(size_t)(mtile * 16 + m) * COUTX + n] = acc[j][r] + bv[j];
            }
        }
    }
}

extern "C" void kernel_launch(void* const* d_in, const int* in_sizes, int n_in,
                              void* d_out, int out_size, void* d_ws, size_t ws_size,
                              hipStream_t stream) {
    (void)in_sizes; (void)n_in; (void)out_size; (void)d_ws; (void)ws_size;
    const float* x    = (const float*)d_in[0];
    const float* kern = (const float*)d_in[1];
    const float* bias = (const float*)d_in[2];
    float* out        = (float*)d_out;

    dim3 grid(MTILES / MT_PER_BLOCK, NG, 1);   // 196 x 8 blocks
    group_conv_wmma_bf16<<<grid, 128, 0, stream>>>(x, kern, bias, out);
}
